// Encoder_30537217475005
// MI455X (gfx1250) — matrix operand order repaired and checked by reading, not compiled, not run on hardware
//
#include <hip/hip_runtime.h>
#include <string.h>

// ---------------------------------------------------------------------------
// GRU encoder for MI455X (gfx1250, wave32, WMMA bf16 16x16x32 + TDM staging)
//   B=128, S=512, V=15000, E=300 (padded to 320), H=1024, 3H=3072
// ---------------------------------------------------------------------------

#define B_   128
#define S_   512
#define V_   15000
#define E_   300
#define EP   320            // E padded to multiple of 32 (zero fill)
#define H_   1024
#define G3   3072           // 3*H
#define LDSTRIDE 1040       // LDS row stride in elements (hw pad: 512dw + 2x4dw)

typedef __attribute__((ext_vector_type(16))) __bf16        v16bf;
typedef __attribute__((ext_vector_type(8)))  float         v8f;
typedef __attribute__((ext_vector_type(4)))  unsigned int  u32x4;
typedef __attribute__((ext_vector_type(8)))  unsigned int  u32x8;

union AB32 { u32x4 q[2]; v16bf v; };

// round-to-nearest-even fp32 -> bf16 (raw u16)
static __device__ __forceinline__ unsigned short f2bf_rne(float f) {
  union { float f; unsigned u; } a; a.f = f;
  unsigned u = a.u;
  if ((u & 0x7fffffffu) > 0x7f800000u) return (unsigned short)((u >> 16) | 0x0040u);
  return (unsigned short)((u + 0x7fffu + ((u >> 16) & 1u)) >> 16);
}

// Assemble a 32-byte WMMA operand from two 16B chunks.
static __device__ __forceinline__ v16bf load_ab(const unsigned short* p0,
                                                const unsigned short* p1) {
  AB32 u;
  u.q[0] = *(const u32x4*)p0;
  u.q[1] = *(const u32x4*)p1;
  return u.v;
}
static __device__ __forceinline__ v16bf load_ab_lds(const unsigned short* p) {
  AB32 u;
  u.q[0] = *(const u32x4*)p;
  u.q[1] = *(const u32x4*)(p + 8);
  return u.v;
}

static __device__ __forceinline__ v8f wmma_bf16(v16bf a, v16bf b, v8f c) {
  return __builtin_amdgcn_wmma_f32_16x16x32_bf16(false, a, false, b,
                                                 (short)0, c, false, false);
}

// ---------------------------------------------------------------------------
// Kernel 0: repack fp32 weights -> bf16 (K zero-padded), zero h0, reset barrier
// ---------------------------------------------------------------------------
__global__ void __launch_bounds__(256) pack_kernel(
    const float* __restrict__ emb,   // [V][E]
    const float* __restrict__ w_ih,  // [3H][E]
    const float* __restrict__ w_hh,  // [3H][H]
    unsigned short* __restrict__ embb,  // [V][EP]
    unsigned short* __restrict__ wihb,  // [3H][EP]
    unsigned short* __restrict__ whhb,  // [3H][H]
    unsigned short* __restrict__ hbf,   // ping-pong [2][B][H] (buf0 zeroed)
    float*          __restrict__ hf,    // ping-pong [2][B][H] (buf0 zeroed)
    unsigned int*   __restrict__ bar)
{
  const size_t tid    = (size_t)blockIdx.x * blockDim.x + threadIdx.x;
  const size_t stride = (size_t)gridDim.x * blockDim.x;

  for (size_t i = tid; i < (size_t)V_ * EP; i += stride) {
    size_t r = i / EP, c = i % EP;
    float v = (c < E_) ? emb[r * E_ + c] : 0.0f;
    embb[i] = f2bf_rne(v);
  }
  for (size_t i = tid; i < (size_t)G3 * EP; i += stride) {
    size_t r = i / EP, c = i % EP;
    float v = (c < E_) ? w_ih[r * E_ + c] : 0.0f;
    wihb[i] = f2bf_rne(v);
  }
  for (size_t i = tid; i < (size_t)G3 * H_; i += stride)
    whhb[i] = f2bf_rne(w_hh[i]);
  for (size_t i = tid; i < (size_t)B_ * H_; i += stride) {
    hbf[i] = 0;
    hf[i]  = 0.0f;
  }
  if (tid == 0) *bar = 0u;
}

// ---------------------------------------------------------------------------
// Kernel 1: xg[s][b][g] = sum_k emb_bf16[x[b][s]][k] * w_ih_bf16[g][k] + b_ih[g]
//   grid = (S, 3H/64); block = 256 (8 waves). Wave w handles rows b = 16w..16w+15,
//   4 WMMA n-subtiles of 16 cols. K loop: EP/32 = 10 WMMA k-steps.
// ---------------------------------------------------------------------------
__global__ void __launch_bounds__(256) embed_gemm_kernel(
    const int* __restrict__ x,                   // [B][S]
    const unsigned short* __restrict__ embb,     // [V][EP]
    const unsigned short* __restrict__ wihb,     // [3H][EP]
    const float* __restrict__ b_ih,              // [3H]
    float* __restrict__ xg)                      // [S][B][3H]
{
  const int lane = threadIdx.x & 31;
  const int wave = threadIdx.x >> 5;
  const int s    = blockIdx.x;
  const int n0   = blockIdx.y * 64;
  const int col  = lane & 15;
  const int khi  = lane >> 4;           // which k-half this lane holds

  // gathered A row for this lane (token embedding, bf16, padded)
  const int brow = wave * 16 + col;
  const int tok  = x[brow * S_ + s];
  const unsigned short* arow = embb + (size_t)tok * EP;

  v8f c0 = {}; v8f c1 = {}; v8f c2 = {}; v8f c3 = {};

  for (int kk = 0; kk < EP; kk += 32) {
    const int ka = kk + khi * 16;                // A: lanes 0-15 K=0..15
    v16bf a = load_ab(arow + ka, arow + ka + 8);

    const int kb = kk + khi * 16;                // B: lanes 0-15 K=0..15
    const unsigned short* bp = wihb + (size_t)(n0 + col) * EP + kb;
    v16bf b0 = load_ab(bp + (size_t)0 * 16 * EP, bp + (size_t)0 * 16 * EP + 8);
    v16bf b1 = load_ab(bp + (size_t)1 * 16 * EP, bp + (size_t)1 * 16 * EP + 8);
    v16bf b2 = load_ab(bp + (size_t)2 * 16 * EP, bp + (size_t)2 * 16 * EP + 8);
    v16bf b3 = load_ab(bp + (size_t)3 * 16 * EP, bp + (size_t)3 * 16 * EP + 8);

    c0 = wmma_bf16(a, b0, c0);
    c1 = wmma_bf16(a, b1, c1);
    c2 = wmma_bf16(a, b2, c2);
    c3 = wmma_bf16(a, b3, c3);
  }

  // C layout: VGPR r, lanes 0-15 -> M=r ; lanes 16-31 -> M=r+8 ; N = lane%16
#pragma unroll
  for (int r = 0; r < 8; ++r) {
    const int bm   = wave * 16 + r + 8 * khi;
    const size_t o = ((size_t)s * B_ + bm) * G3;
    const int g0 = n0 + 0  + col;
    const int g1 = n0 + 16 + col;
    const int g2 = n0 + 32 + col;
    const int g3 = n0 + 48 + col;
    xg[o + g0] = c0[r] + b_ih[g0];
    xg[o + g1] = c1[r] + b_ih[g1];
    xg[o + g2] = c2[r] + b_ih[g2];
    xg[o + g3] = c3[r] + b_ih[g3];
  }
}

// ---------------------------------------------------------------------------
// Kernel 2: persistent GRU recurrence.
//   grid = 64 workgroups (one per 16-col j-tile of H), 256 threads = 8 waves.
//   w_hh slice [3 gates][16 rows][1024 k] bf16 is DMA'd into LDS by ONE
//   tensor_load_to_lds (TDM): 3D tile, dim0=1024, dim1=16 (stride H),
//   dim2=3 (stride H*H). HW pad (interval=256dw, amount=4dw) gives a
//   1040-element LDS row stride + a 4dw mid-row gap at k=512 (bank spread,
//   16B alignment preserved).
// ---------------------------------------------------------------------------
__global__ void __launch_bounds__(256) gru_kernel(
    const float* __restrict__ xg,             // [S][B][3H] (b_ih folded in)
    const unsigned short* __restrict__ whhb,  // [3H][H] bf16
    const float* __restrict__ b_hh,           // [3H]
    unsigned short* __restrict__ hbf,         // [2][B][H] bf16 ping-pong
    float*          __restrict__ hf,          // [2][B][H] fp32 ping-pong
    float*          __restrict__ out,         // [B][H]
    unsigned int*   __restrict__ bar)
{
  extern __shared__ unsigned short lds_w[];   // [48][LDSTRIDE] via TDM pad

  const int lane = threadIdx.x & 31;
  const int wave = threadIdx.x >> 5;
  const int j0   = blockIdx.x * 16;           // this WG's 16 columns of H
  const int col  = lane & 15;
  const int khi  = lane >> 4;
  const int b0   = wave * 16;

  // ---- TDM: DMA w_hh[q*H + j0 + r][0..1023] (q=0..2, r=0..15) into LDS ----
  if (wave == 0) {
    const unsigned long long ga =
        (unsigned long long)(uintptr_t)(whhb + (size_t)j0 * H_);

    u32x4 g0;
    g0[0] = 1u;                                    // count=1 (valid user D#)
    g0[1] = 0u;                                    // lds_addr = 0 (dynamic LDS base)
    g0[2] = (unsigned)(ga & 0xffffffffu);          // global_addr[31:0]
    g0[3] = (unsigned)((ga >> 32) & 0x01ffffffu)   // global_addr[56:32]
          | (2u << 30);                            // type = 2 (image)

    u32x8 g1;
    g1[0] = (1u << 16)      // data_size = 1 (2 bytes)
          | (1u << 20)      // pad_enable
          | (7u << 22)      // pad_interval: 256 DWORDs
          | (3u << 25);     // pad_amount:   4 DWORDs
    g1[1] = (1024u & 0xffffu) << 16;               // tensor_dim0[15:0] (=1024)
    g1[2] = (1024u >> 16) | ((0x100000u & 0xffffu) << 16); // td0 hi | tensor_dim1 lo
    g1[3] = (0x100000u >> 16) | (1024u << 16);     // tensor_dim1 hi | tile_dim0=1024
    g1[4] = 16u | (3u << 16);                      // tile_dim1=16, tile_dim2=3
    g1[5] = 1024u;                                 // tensor_dim0_stride lo (=H)
    g1[6] = 0u | (((1024u * 1024u) & 0xffffu) << 16); // td0s hi | td1_stride lo
    g1[7] = (1024u * 1024u) >> 16;                 // tensor_dim1_stride hi (=H*H)

    u32x4 g2;
    g2[0] = 16u;   // tensor_dim2 (>= 3, no OOB clip)
    g2[1] = 0u;    // tensor_dim3
    g2[2] = 0u;    // tensor_dim2_stride lo
    g2[3] = 0u;    // stride hi | tile_dim3 = 0

    u32x4 g3 = { 0u, 0u, 0u, 0u };

    asm volatile("tensor_load_to_lds %0, %1, %2, %3"
                 :: "s"(g0), "s"(g1), "s"(g2), "s"(g3)
                 : "memory");
    __builtin_amdgcn_s_wait_tensorcnt(0);
  }
  __syncthreads();

  const float bhr = b_hh[j0 + col];
  const float bhz = b_hh[H_ + j0 + col];
  const float bhn = b_hh[2 * H_ + j0 + col];

  for (int s = 0; s < S_; ++s) {
    const int cur = s & 1, nxt = cur ^ 1;
    const unsigned short* hread = hbf + (size_t)cur * B_ * H_;
    const float*          hfr   = hf  + (size_t)cur * B_ * H_;
    unsigned short*       hw16  = hbf + (size_t)nxt * B_ * H_;
    float*                hw32  = hf  + (size_t)nxt * B_ * H_;

    v8f cr = {}; v8f cz = {}; v8f cn = {};

    const unsigned short* arow = hread + (size_t)(b0 + col) * H_;
#pragma unroll 4
    for (int kk = 0; kk < H_; kk += 32) {
      const int ka = kk + khi * 16;
      v16bf a = load_ab(arow + ka, arow + ka + 8);

      const int kb   = kk + khi * 16;
      const int kpad = kb + ((kb >> 9) << 3);   // +8 elems past TDM mid-row pad
      v16bf br = load_ab_lds(lds_w + (0 * 16 + col) * LDSTRIDE + kpad);
      v16bf bz = load_ab_lds(lds_w + (1 * 16 + col) * LDSTRIDE + kpad);
      v16bf bn = load_ab_lds(lds_w + (2 * 16 + col) * LDSTRIDE + kpad);

      cr = wmma_bf16(a, br, cr);
      cz = wmma_bf16(a, bz, cz);
      cn = wmma_bf16(a, bn, cn);
    }

    // fused GRU gates on the f32 accumulators
    const int j = j0 + col;
#pragma unroll
    for (int r = 0; r < 8; ++r) {
      const int brow = b0 + r + 8 * khi;
      const size_t xb = ((size_t)s * B_ + brow) * G3;
      const float xr = xg[xb + j];
      const float xz = xg[xb + H_ + j];
      const float xn = xg[xb + 2 * H_ + j];

      const float rg = 1.0f / (1.0f + __expf(-(xr + cr[r] + bhr)));
      const float zg = 1.0f / (1.0f + __expf(-(xz + cz[r] + bhz)));
      const float ng = tanhf(xn + rg * (cn[r] + bhn));

      const float hp = hfr[(size_t)brow * H_ + j];
      const float hn2 = (1.0f - zg) * ng + zg * hp;

      hw32[(size_t)brow * H_ + j] = hn2;
      hw16[(size_t)brow * H_ + j] = f2bf_rne(hn2);
      if (s == S_ - 1) out[(size_t)brow * H_ + j] = hn2;
    }

    // prefetch next step's xg slice for this lane's rows (global_prefetch_b8)
    if (s + 1 < S_)
      __builtin_prefetch(&xg[((size_t)(s + 1) * B_ + b0 + col) * G3 + j0], 0, 1);

    // ---- device-wide barrier: monotonically increasing arrival counter
    __threadfence();
    __syncthreads();
    if (threadIdx.x == 0) {
      __hip_atomic_fetch_add(bar, 1u, __ATOMIC_RELEASE, __HIP_MEMORY_SCOPE_AGENT);
      const unsigned target = (unsigned)gridDim.x * (unsigned)(s + 1);
      while (__hip_atomic_load(bar, __ATOMIC_ACQUIRE, __HIP_MEMORY_SCOPE_AGENT)
             < target) {
        __builtin_amdgcn_s_sleep(1);
      }
    }
    __syncthreads();
  }
}

// ---------------------------------------------------------------------------
// Host launcher
// ---------------------------------------------------------------------------
extern "C" void kernel_launch(void* const* d_in, const int* in_sizes, int n_in,
                              void* d_out, int out_size, void* d_ws, size_t ws_size,
                              hipStream_t stream) {
  (void)in_sizes; (void)n_in; (void)out_size; (void)ws_size;

  const int*   x    = (const int*)  d_in[0];   // [B][S]
  const float* emb  = (const float*)d_in[1];   // [V][E]
  const float* w_ih = (const float*)d_in[2];   // [3H][E]
  const float* w_hh = (const float*)d_in[3];   // [3H][H]
  const float* b_ih = (const float*)d_in[4];   // [3H]
  const float* b_hh = (const float*)d_in[5];   // [3H]
  float*       out  = (float*)d_out;           // [B][H]

  // workspace carving (256B aligned)
  char* ws = (char*)d_ws;
  size_t o = 0;
  auto carve = [&](size_t bytes) {
    char* p = ws + o;
    o += (bytes + 255) & ~(size_t)255;
    return p;
  };
  unsigned short* embb = (unsigned short*)carve((size_t)V_ * EP * 2);
  unsigned short* wihb = (unsigned short*)carve((size_t)G3 * EP * 2);
  unsigned short* whhb = (unsigned short*)carve((size_t)G3 * H_ * 2);
  unsigned short* hbf  = (unsigned short*)carve((size_t)2 * B_ * H_ * 2);
  float*          hf   = (float*)         carve((size_t)2 * B_ * H_ * 4);
  unsigned int*   bar  = (unsigned int*)  carve(256);
  float*          xg   = (float*)         carve((size_t)S_ * B_ * G3 * 4);

  // 0) repack weights to bf16, zero h0, reset barrier
  pack_kernel<<<2048, 256, 0, stream>>>(emb, w_ih, w_hh,
                                        embb, wihb, whhb, hbf, hf, bar);

  // 1) embedding gather + input projection (WMMA bf16)
  embed_gemm_kernel<<<dim3(S_, G3 / 64), 256, 0, stream>>>(x, embb, wihb, b_ih, xg);

  // 2) persistent GRU recurrence (dynamic LDS: 48 x 520 DWORDs = 99,840 B)
  const int ldsBytes = 48 * (LDSTRIDE / 2) * 4;
  (void)hipFuncSetAttribute((const void*)gru_kernel,
                            hipFuncAttributeMaxDynamicSharedMemorySize, ldsBytes);
  gru_kernel<<<H_ / 16, 256, ldsBytes, stream>>>(xg, whhb, b_hh, hbf, hf, out, bar);
}
